// DualGNN_10746008175453
// MI455X (gfx1250) — matrix operand order repaired and checked by reading, not compiled, not run on hardware
//
#include <hip/hip_runtime.h>
#include <hip/hip_bf16.h>

// ---------------------------------------------------------------------------
// DualGNN forward on gfx1250 (MI455X).  MLPs use v_wmma_f32_16x16x32_bf16
// with zero-padded bf16 staged operands (vectorized b128 loads, no per-lane
// predication in the K loop).  Graph ops are atomic scatter-adds (L2-bound).
// wave32 throughout.
// ---------------------------------------------------------------------------

#define NUM_USER 30000
#define NUM_ITEM 30000
#define NTOT     (NUM_USER + NUM_ITEM)
#define DIM      64
#define HID      256
#define NEDGE    1000000   // 2 * E_PAIRS directed edges
#define KNN      10
#define BATCH    2048

typedef __attribute__((ext_vector_type(16))) __bf16 v16bf;
typedef __attribute__((ext_vector_type(8)))  __bf16 v8bf;
typedef __attribute__((ext_vector_type(8)))  float  v8f;

union V16U { v16bf v; v8bf h[2]; };

// ---------------------------------------------------------------------------
// Convert f32 -> bf16 with zero padding of the K dimension to Kpad.
// out is (rows x Kpad) bf16 row-major; in is (rows x K) f32 row-major.
// ---------------------------------------------------------------------------
__global__ void cvt_pad_bf16_kernel(const float* __restrict__ in, __bf16* __restrict__ out,
                                    int K, int Kpad, long long total)
{
    long long i = (long long)blockIdx.x * blockDim.x + threadIdx.x;
    if (i >= total) return;
    long long r = i / Kpad;
    int kk = (int)(i - r * Kpad);
    out[i] = (kk < K) ? (__bf16)in[r * K + kk] : (__bf16)0.0f;
}

// ---------------------------------------------------------------------------
// WMMA GEMM:  Out(M x Nout) = act( A(M x K) * W(Nout x K)^T + bias )
// A, W are bf16 row-major with padded leading dim lda (multiple of 32).
// One wave computes a 16 x (16*NT) strip (A fragment reused NT times).
// 4 waves per block span 64*NT output columns.  M must be a multiple of 16.
// ---------------------------------------------------------------------------
template <int NT, typename TO, bool ACT>
__global__ __launch_bounds__(128)
void gemm_bf16_wmma_kernel(const __bf16* __restrict__ A, int lda,
                           const __bf16* __restrict__ W,
                           const float* __restrict__ bias,
                           TO* __restrict__ Out, int ldo, int K)
{
    const int lane = threadIdx.x & 31;
    const int wave = threadIdx.x >> 5;
    const int half = lane >> 4;       // which 16-lane half of the wave
    const int l16  = lane & 15;

    const int tileM  = blockIdx.x * 16;
    const int tileN0 = (blockIdx.y * 4 + wave) * (16 * NT);

    // A 16x32 bf16 fragment (ISA 7.12.2): lane supplies row tileM+l16,
    // two contiguous 8-element runs at k0+8*half and k0+16+8*half.
    const v8bf* Arow = (const v8bf*)(A + (size_t)(tileM + l16) * lda);
    // B 32x16 fragment: lane supplies column tileN+l16 (= W row),
    // two contiguous 8-element runs at k0+8*half and k0+16+8*half.
    const v8bf* Brow[NT];
#pragma unroll
    for (int t = 0; t < NT; ++t)
        Brow[t] = (const v8bf*)(W + (size_t)(tileN0 + t * 16 + l16) * lda);

    v8f c[NT];
#pragma unroll
    for (int t = 0; t < NT; ++t) c[t] = (v8f){0,0,0,0,0,0,0,0};

    for (int k0 = 0; k0 < K; k0 += 32) {
        const int i8 = k0 >> 3;      // index in 8-element units
        V16U a;
        a.h[0] = Arow[i8 + half];
        a.h[1] = Arow[i8 + half + 2];
#pragma unroll
        for (int t = 0; t < NT; ++t) {
            V16U b;
            b.h[0] = Brow[t][i8 + half];
            b.h[1] = Brow[t][i8 + half + 2];
            c[t] = __builtin_amdgcn_wmma_f32_16x16x32_bf16(
                       /*neg_a=*/false, a.v, /*neg_b=*/false, b.v,
                       /*c_mod=*/(short)0, c[t],
                       /*reuse_a=*/false, /*reuse_b=*/false);
        }
    }

    // Epilogue: c[t][g] -> Out[tileM + g + 8*half][tileN0 + 16*t + l16]
#pragma unroll
    for (int t = 0; t < NT; ++t) {
#pragma unroll
        for (int g = 0; g < 8; ++g) {
            int row = tileM + g + 8 * half;
            int col = tileN0 + t * 16 + l16;
            float v = c[t][g] + bias[col];
            if (ACT) v = (v > 0.0f) ? v : 0.01f * v;   // jax leaky_relu slope
            Out[(size_t)row * ldo + col] = (TO)v;
        }
    }
}

// ---------------------------------------------------------------------------
// Degree / dinv
// ---------------------------------------------------------------------------
__global__ void degree_kernel(const int* __restrict__ src, float* __restrict__ deg, int E)
{
    int e = blockIdx.x * blockDim.x + threadIdx.x;
    if (e < E) atomicAdd(&deg[src[e]], 1.0f);
}

__global__ void dinv_kernel(float* __restrict__ deg)
{
    int i = blockIdx.x * blockDim.x + threadIdx.x;
    if (i < NTOT) {
        float d = deg[i];
        deg[i] = (d > 0.0f) ? rsqrtf(d) : 0.0f;
    }
}

// ---------------------------------------------------------------------------
// L2-normalize rows of x.  Users sourced from pref, items in place.
// One 32-lane wave per 64-wide row.
// ---------------------------------------------------------------------------
__global__ void l2norm_kernel(const float* __restrict__ pref, float* __restrict__ x)
{
    int row  = (blockIdx.x * blockDim.x + threadIdx.x) >> 5;
    int lane = threadIdx.x & 31;
    if (row >= NTOT) return;
    const float* s = (row < NUM_USER) ? (pref + (size_t)row * DIM)
                                      : (x    + (size_t)row * DIM);
    float v0 = s[lane], v1 = s[lane + 32];
    float ss = v0 * v0 + v1 * v1;
#pragma unroll
    for (int m = 16; m >= 1; m >>= 1) ss += __shfl_xor(ss, m, 32);
    float inv = 1.0f / fmaxf(sqrtf(ss), 1e-12f);
    x[(size_t)row * DIM + lane]      = v0 * inv;
    x[(size_t)row * DIM + 32 + lane] = v1 * inv;
}

// ---------------------------------------------------------------------------
// GCN conv: out[dst] += dinv[src]*dinv[dst] * x[src]  (one thread per
// (edge, dim); coalesced f32 atomics -> global_atomic_add_f32, L2-resident)
// ---------------------------------------------------------------------------
__global__ void gcn_conv_kernel(const int* __restrict__ src, const int* __restrict__ dst,
                                const float* __restrict__ dinv,
                                const float* __restrict__ xin, float* __restrict__ out, int E)
{
    long long idx = (long long)blockIdx.x * blockDim.x + threadIdx.x;
    if (idx >= (long long)E * DIM) return;
    int e = (int)(idx >> 6);
    int d = (int)(idx & 63);
    int s = src[e];
    int t = dst[e];
    float w = dinv[s] * dinv[t];
    atomicAdd(&out[(size_t)t * DIM + d], w * xin[(size_t)s * DIM + d]);
}

// ---------------------------------------------------------------------------
// rep = h + x + h1 ; repsum += rep/3 ; user rows cached per-modality
// ---------------------------------------------------------------------------
__global__ void rep_combine_kernel(const float* __restrict__ x, const float* __restrict__ h,
                                   const float* __restrict__ h1, float* __restrict__ repsum,
                                   float* __restrict__ urep)
{
    long long i = (long long)blockIdx.x * blockDim.x + threadIdx.x;
    if (i >= (long long)NTOT * DIM) return;
    float r = h[i] + x[i] + h1[i];
    repsum[i] += r * (1.0f / 3.0f);
    if (i < (long long)NUM_USER * DIM) urep[i] = r;
}

// ---------------------------------------------------------------------------
// user base: weight_u-mixed modality reps
// ---------------------------------------------------------------------------
__global__ void user_base_kernel(const float* __restrict__ u0, const float* __restrict__ u1,
                                 const float* __restrict__ u2, const float* __restrict__ wu,
                                 float* __restrict__ ubase)
{
    int i = blockIdx.x * blockDim.x + threadIdx.x;
    if (i >= NUM_USER * DIM) return;
    int u = i >> 6;
    ubase[i] = wu[u * 3 + 0] * u0[i] + wu[u * 3 + 1] * u1[i] + wu[u * 3 + 2] * u2[i];
}

// ---------------------------------------------------------------------------
// result: users = ubase + KNN-weighted gather of ubase ; items = repsum
// ---------------------------------------------------------------------------
__global__ void result_kernel(const float* __restrict__ ubase, const int* __restrict__ graph,
                              const float* __restrict__ uwm, const float* __restrict__ repsum,
                              float* __restrict__ result)
{
    long long i = (long long)blockIdx.x * blockDim.x + threadIdx.x;
    if (i >= (long long)NTOT * DIM) return;
    int u = (int)(i >> 6);
    if (u < NUM_USER) {
        int d = (int)(i & 63);
        float acc = ubase[i];
#pragma unroll
        for (int k = 0; k < KNN; ++k) {
            int g = graph[u * KNN + k];
            acc += uwm[u * KNN + k] * ubase[(size_t)g * DIM + d];
        }
        result[i] = acc;
    } else {
        result[i] = repsum[i];
    }
}

// ---------------------------------------------------------------------------
// scoring: one wave per (pos|neg, batch) dot product
// ---------------------------------------------------------------------------
__global__ void score_kernel(const int* __restrict__ un, const int* __restrict__ pn,
                             const int* __restrict__ nn, const float* __restrict__ result,
                             float* __restrict__ out)
{
    int p    = (blockIdx.x * blockDim.x + threadIdx.x) >> 5;
    int lane = threadIdx.x & 31;
    if (p >= 2 * BATCH) return;
    int b = (p < BATCH) ? p : p - BATCH;
    int i = un[b];
    int j = (p < BATCH) ? pn[b] : nn[b];
    const float* ri = result + (size_t)i * DIM;
    const float* rj = result + (size_t)j * DIM;
    float s = ri[lane] * rj[lane] + ri[lane + 32] * rj[lane + 32];
#pragma unroll
    for (int m = 16; m >= 1; m >>= 1) s += __shfl_xor(s, m, 32);
    if (lane == 0) out[p] = s;
}

// ---------------------------------------------------------------------------
extern "C" void kernel_launch(void* const* d_in, const int* in_sizes, int n_in,
                              void* d_out, int out_size, void* d_ws, size_t ws_size,
                              hipStream_t stream)
{
    const int*   user_nodes = (const int*)d_in[0];
    const int*   pos_nodes  = (const int*)d_in[1];
    const int*   neg_nodes  = (const int*)d_in[2];
    const int*   user_graph = (const int*)d_in[3];
    const int*   edge_index = (const int*)d_in[4];
    const float* uwm        = (const float*)d_in[5];
    const float* feat[3] = {(const float*)d_in[6],  (const float*)d_in[7],  (const float*)d_in[8]};
    const float* pref[3] = {(const float*)d_in[9],  (const float*)d_in[10], (const float*)d_in[11]};
    const float* W1[3]   = {(const float*)d_in[12], (const float*)d_in[16], (const float*)d_in[20]};
    const float* b1[3]   = {(const float*)d_in[13], (const float*)d_in[17], (const float*)d_in[21]};
    const float* W2[3]   = {(const float*)d_in[14], (const float*)d_in[18], (const float*)d_in[22]};
    const float* b2[3]   = {(const float*)d_in[15], (const float*)d_in[19], (const float*)d_in[23]};
    const float* weight_u = (const float*)d_in[24];
    const int Dm[3]   = {1024, 128, 100};
    const int Kpad[3] = {1024, 128, 128};   // K padded to multiple of 32

    const int* esrc = edge_index;
    const int* edst = edge_index + NEDGE;

    // workspace layout (float units; all bf16 regions 16B-aligned)
    float* ws = (float*)d_ws;
    size_t o = 0;
    float*  dinv   = ws + o; o += NTOT;
    float*  xb     = ws + o; o += (size_t)NTOT * DIM;
    float*  hb     = ws + o; o += (size_t)NTOT * DIM;
    float*  h1b    = ws + o; o += (size_t)NTOT * DIM;
    float*  repsum = ws + o; o += (size_t)NTOT * DIM;
    float*  urep   = ws + o; o += (size_t)3 * NUM_USER * DIM;
    __bf16* hidbf  = (__bf16*)(ws + o); o += (size_t)NUM_ITEM * HID / 2;
    __bf16* featbf = (__bf16*)(ws + o); o += (size_t)NUM_ITEM * 1024 / 2;  // max Kpad
    __bf16* w1bf   = (__bf16*)(ws + o); o += (size_t)HID * 1024 / 2;
    __bf16* w2bf   = (__bf16*)(ws + o); o += (size_t)DIM * HID / 2;
    // dead-after-loop aliases
    float* ubase   = hb;    // user base rep (used after modality loop)
    float* resultb = h1b;   // final result matrix (used after modality loop)

    const int TPB = 256;

    // degrees -> dinv
    hipMemsetAsync(dinv, 0, NTOT * sizeof(float), stream);
    degree_kernel<<<(NEDGE + TPB - 1) / TPB, TPB, 0, stream>>>(esrc, dinv, NEDGE);
    dinv_kernel<<<(NTOT + TPB - 1) / TPB, TPB, 0, stream>>>(dinv);

    hipMemsetAsync(repsum, 0, (size_t)NTOT * DIM * sizeof(float), stream);

    const long long nd   = (long long)NTOT * DIM;
    const long long edim = (long long)NEDGE * DIM;

    for (int m = 0; m < 3; ++m) {
        // stage bf16 operands (zero-padded K)
        long long tf = (long long)NUM_ITEM * Kpad[m];
        cvt_pad_bf16_kernel<<<(unsigned)((tf + TPB - 1) / TPB), TPB, 0, stream>>>(
            feat[m], featbf, Dm[m], Kpad[m], tf);
        long long tw1 = (long long)HID * Kpad[m];
        cvt_pad_bf16_kernel<<<(unsigned)((tw1 + TPB - 1) / TPB), TPB, 0, stream>>>(
            W1[m], w1bf, Dm[m], Kpad[m], tw1);
        long long tw2 = (long long)DIM * HID;
        cvt_pad_bf16_kernel<<<(unsigned)((tw2 + TPB - 1) / TPB), TPB, 0, stream>>>(
            W2[m], w2bf, HID, HID, tw2);

        // MLP layer 1: hidden = leaky_relu(feat @ W1^T + b1)  (16x256 per block)
        gemm_bf16_wmma_kernel<4, __bf16, true><<<dim3(NUM_ITEM / 16, 1), 128, 0, stream>>>(
            featbf, Kpad[m], w1bf, b1[m], hidbf, HID, Kpad[m]);
        // MLP layer 2: temp = hidden @ W2^T + b2 -> item rows of x (16x64 per block)
        gemm_bf16_wmma_kernel<1, float, false><<<dim3(NUM_ITEM / 16, 1), 128, 0, stream>>>(
            hidbf, HID, w2bf, b2[m], xb + (size_t)NUM_USER * DIM, DIM, HID);

        // x = l2norm(concat(pref, temp))
        l2norm_kernel<<<(NTOT * 32 + TPB - 1) / TPB, TPB, 0, stream>>>(pref[m], xb);
        // h = conv(x)
        hipMemsetAsync(hb, 0, (size_t)NTOT * DIM * sizeof(float), stream);
        gcn_conv_kernel<<<(unsigned)((edim + TPB - 1) / TPB), TPB, 0, stream>>>(
            esrc, edst, dinv, xb, hb, NEDGE);
        // h1 = conv(h)
        hipMemsetAsync(h1b, 0, (size_t)NTOT * DIM * sizeof(float), stream);
        gcn_conv_kernel<<<(unsigned)((edim + TPB - 1) / TPB), TPB, 0, stream>>>(
            esrc, edst, dinv, hb, h1b, NEDGE);
        // rep = h + x + h1 ; accumulate
        rep_combine_kernel<<<(unsigned)((nd + TPB - 1) / TPB), TPB, 0, stream>>>(
            xb, hb, h1b, repsum, urep + (size_t)m * NUM_USER * DIM);
    }

    user_base_kernel<<<(NUM_USER * DIM + TPB - 1) / TPB, TPB, 0, stream>>>(
        urep, urep + (size_t)NUM_USER * DIM, urep + (size_t)2 * NUM_USER * DIM,
        weight_u, ubase);
    result_kernel<<<(unsigned)((nd + TPB - 1) / TPB), TPB, 0, stream>>>(
        ubase, user_graph, uwm, repsum, resultb);
    score_kernel<<<(2 * BATCH * 32 + TPB - 1) / TPB, TPB, 0, stream>>>(
        user_nodes, pos_nodes, neg_nodes, resultb, (float*)d_out);
}
